// TreeConvLSTMCell3d_5059471474866
// MI455X (gfx1250) — matrix operand order repaired and checked by reading, not compiled, not run on hardware
//
#include <hip/hip_runtime.h>

typedef __attribute__((ext_vector_type(16))) _Float16 v16h;
typedef __attribute__((ext_vector_type(8)))  _Float16 v8h;
typedef __attribute__((ext_vector_type(8)))  float    v8f;

union HFrag { v16h v; v8h h[2]; _Float16 e[16]; };

constexpr int C16     = 16;
constexpr int SPAT    = 4096;   // 16^3
constexpr int KCH     = 8;      // children
constexpr int NB      = 128;    // batch
constexpr int TAPS_P  = 28;     // 27 real taps + 1 zero-weight pad tap

// ---------------------------------------------------------------------------
// One 16(oc) x 16(w) output tile of the 3x3x3 conv via 14 WMMA f16 ops.
// A layout (16-bit 16x32): lane L -> row M=L%16; half=L/16 selects tap of the
//   pair (tap0+half); elements 0..15 = ic 0..15 -> two 16B LDS reads.
// B layout (16-bit 32x16): lane L -> col N=L%16; half selects tap of the pair;
//   dword r holds ic pair (2r,2r+1) -> one site's 16-ch vector (2x b128).
// ---------------------------------------------------------------------------
__device__ __forceinline__
v8f conv_tile(const _Float16* s_in, const _Float16* s_w, int Noc,
              int ocBase, int dloc, int hrow, int lane, v8f acc) {
    const int m    = lane & 15;   // A row (oc) and B col (w)
    const int half = lane >> 4;   // K half
#pragma unroll
    for (int tp = 0; tp < 14; ++tp) {
        const int tap0 = tp * 2;
        HFrag a, b;
        const _Float16* wp = s_w + (((tap0 + half) * Noc + ocBase + m) * 16);
        a.h[0] = *(const v8h*)(wp);
        a.h[1] = *(const v8h*)(wp + 8);              // tap0+half (pad tap = zeros)
        int tb = tap0 + half; if (tb > 26) tb = 26;  // clamp pad tap address (A=0 there)
        const int kd = tb / 9, kh = (tb / 3) % 3, kw = tb % 3;
        const _Float16* ip =
            s_in + (((dloc + kd) * 18 + (hrow + kh)) * 18 + (m + kw)) * 16;
        b.h[0] = *(const v8h*)(ip);
        b.h[1] = *(const v8h*)(ip + 8);
        acc = __builtin_amdgcn_wmma_f32_16x16x32_f16(
            false, a.v, false, b.v, (short)0, acc, false, false);
    }
    return acc;
}

// ---------------------------------------------------------------------------
// Kernel A (fused): per (n, 4-deep d-slab), loop k children:
//   - stage mailbox_h[n,k] slab fp32 -> f16 padded LDS (halo zeroed once)
//   - while staging, accumulate h_child = sum_k over the 4 interior slices
//   - f = sigmoid(WMMA conv), c_sum += f * mailbox_c (register accumulators)
// Writes: h_child (f16 site-major) and c_sum (fp32) to workspace.
// 512 threads = 16 waves; 64 row-tiles -> 4 tiles/wave (32 csum VGPRs).
// ---------------------------------------------------------------------------
__global__ void __launch_bounds__(512)
k_fconv_csum(const float* __restrict__ mh, const float* __restrict__ mc,
             const float* __restrict__ ufw, float* __restrict__ csum_ws,
             _Float16* __restrict__ hchild) {
    __shared__ __align__(16) _Float16 s_in[6 * 18 * 18 * 16];  // padded slab, f16
    __shared__ __align__(16) _Float16 s_w[TAPS_P * 16 * 16];
    const int n    = blockIdx.x >> 2;
    const int d0   = (blockIdx.x & 3) * 4;
    const int tid  = threadIdx.x;
    const int wave = tid >> 5, lane = tid & 31;
    const int m    = lane & 15, half = lane >> 4;
    const int site = tid & 255;          // (h,w) site this thread stages
    const int chalf = tid >> 8;          // which 8 channels it stages
    const int hh = site >> 4, ww = site & 15;

    // stage weights [tap][oc][ic], pad tap 27 with zeros
    for (int i = tid; i < TAPS_P * 16 * 16; i += 512) {
        const int tap = i >> 8, oc = (i >> 4) & 15, ic = i & 15;
        s_w[i] = (_Float16)((tap < 27) ? ufw[(oc * 16 + ic) * 27 + tap] : 0.f);
    }
    // zero padded volume once; halo stays zero across children
    for (int i = tid; i < 6 * 18 * 18 * 16; i += 512) s_in[i] = (_Float16)0.f;

    v8f csum[4];
#pragma unroll
    for (int t = 0; t < 4; ++t) csum[t] = v8f{};
    float hacc[4][8];                    // h_child partial sums (4 interior d x 8 ch)
#pragma unroll
    for (int dl = 0; dl < 4; ++dl)
#pragma unroll
        for (int cc = 0; cc < 8; ++cc) hacc[dl][cc] = 0.f;

    for (int k = 0; k < KCH; ++k) {
        __syncthreads();  // also orders zero/weight staging before k=0 fill
        const float* src = mh + (size_t)(n * KCH + k) * C16 * SPAT;
#pragma unroll
        for (int dl = 0; dl < 6; ++dl) {            // fill interior d-slices
            const int d = d0 + dl - 1;
            if (d >= 0 && d <= 15) {
#pragma unroll
                for (int cc = 0; cc < 8; ++cc) {
                    const int c = chalf * 8 + cc;
                    float v = src[(size_t)c * SPAT + d * 256 + site];
                    s_in[((dl * 18 + hh + 1) * 18 + (ww + 1)) * 16 + c] = (_Float16)v;
                    if (dl >= 1 && dl <= 4) hacc[dl - 1][cc] += v;  // child-sum
                }
            }
        }
        __syncthreads();
        const float* mck = mc + (size_t)(n * KCH + k) * C16 * SPAT;
#pragma unroll
        for (int t = 0; t < 4; ++t) {               // 64 row-tiles / 16 waves
            const int tile = wave * 4 + t;
            const int dl = tile >> 4, hrow = tile & 15;
            v8f facc = conv_tile(s_in, s_w, 16, 0, dl, hrow, lane, v8f{});
            const int s = (d0 + dl) * 256 + hrow * 16 + m;
#pragma unroll
            for (int v = 0; v < 8; ++v) {
                const int oc = v + half * 8;
                const float cv = mck[(size_t)oc * SPAT + s];
                const float f  = 1.f / (1.f + __expf(-facc[v]));
                csum[t][v] += f * cv;
            }
        }
    }
    // write c_sum (fp32, NCDHW-per-n layout)
    float* dst = csum_ws + (size_t)n * C16 * SPAT;
#pragma unroll
    for (int t = 0; t < 4; ++t) {
        const int tile = wave * 4 + t;
        const int dl = tile >> 4, hrow = tile & 15;
        const int s = (d0 + dl) * 256 + hrow * 16 + m;
#pragma unroll
        for (int v = 0; v < 8; ++v)
            dst[(size_t)(v + half * 8) * SPAT + s] = csum[t][v];
    }
    // write h_child (f16, site-major [n][site][c]); each interior slice owned
    // by exactly one block, each (site, 8ch) by exactly one thread
#pragma unroll
    for (int dl = 0; dl < 4; ++dl) {
        v8h hv;
#pragma unroll
        for (int cc = 0; cc < 8; ++cc) hv[cc] = (_Float16)hacc[dl][cc];
        _Float16* hp = hchild +
            ((size_t)n * SPAT + (d0 + dl) * 256 + site) * 16 + chalf * 8;
        *(v8h*)hp = hv;
    }
}

// ---------------------------------------------------------------------------
// Kernel B: iou = conv(h_child, U_iou) + b_iou; c = sig(i)*tanh(u)+c_sum;
// h = sig(o)*tanh(c). Output = h || c (fp32).
// ---------------------------------------------------------------------------
__global__ void __launch_bounds__(256)
k_iou(const _Float16* __restrict__ hchild, const float* __restrict__ uiw,
      const float* __restrict__ bias, const float* __restrict__ csum_ws,
      float* __restrict__ out) {
    __shared__ __align__(16) _Float16 s_in[6 * 18 * 18 * 16];
    __shared__ __align__(16) _Float16 s_w[TAPS_P * 48 * 16];
    const int n    = blockIdx.x >> 2;
    const int d0   = (blockIdx.x & 3) * 4;
    const int tid  = threadIdx.x;
    const int wave = tid >> 5, lane = tid & 31;
    const int m    = lane & 15, half = lane >> 4;

    for (int i = tid; i < TAPS_P * 48 * 16; i += 256) {
        const int tap = i / (48 * 16), oc = (i >> 4) % 48, ic = i & 15;
        s_w[i] = (_Float16)((tap < 27) ? uiw[(oc * 16 + ic) * 27 + tap] : 0.f);
    }
    for (int i = tid; i < 6 * 18 * 18 * 16; i += 256) s_in[i] = (_Float16)0.f;
    __syncthreads();
    {   // stage h_child slab (site-major f16 -> 32B per site)
        const int hh = tid >> 4, ww = tid & 15;
        for (int dl = 0; dl < 6; ++dl) {
            const int d = d0 + dl - 1;
            if (d < 0 || d > 15) continue;
            const _Float16* sp = hchild + ((size_t)n * SPAT + d * 256 + tid) * 16;
            HFrag f;
            f.h[0] = *(const v8h*)(sp);
            f.h[1] = *(const v8h*)(sp + 8);
            _Float16* dp = &s_in[((dl * 18 + hh + 1) * 18 + (ww + 1)) * 16];
            *(v8h*)(dp)     = f.h[0];
            *(v8h*)(dp + 8) = f.h[1];
        }
    }
    __syncthreads();

    const float* csn = csum_ws + (size_t)n * C16 * SPAT;
    const size_t outHalf = (size_t)NB * C16 * SPAT;
    for (int t = 0; t < 8; ++t) {
        const int tile = wave * 8 + t;
        const int dl = tile >> 4, hrow = tile & 15;
        v8f ai = conv_tile(s_in, s_w, 48, 0,  dl, hrow, lane, v8f{});
        v8f ao = conv_tile(s_in, s_w, 48, 16, dl, hrow, lane, v8f{});
        v8f au = conv_tile(s_in, s_w, 48, 32, dl, hrow, lane, v8f{});
        const int s = (d0 + dl) * 256 + hrow * 16 + m;
#pragma unroll
        for (int v = 0; v < 8; ++v) {
            const int c = v + half * 8;
            const float iv = ai[v] + bias[(size_t)(c)      * SPAT + s];
            const float ov = ao[v] + bias[(size_t)(16 + c) * SPAT + s];
            const float uv = au[v] + bias[(size_t)(32 + c) * SPAT + s];
            const float cs = csn[(size_t)c * SPAT + s];
            const float cval = (1.f / (1.f + __expf(-iv))) * tanhf(uv) + cs;
            const float hval = (1.f / (1.f + __expf(-ov))) * tanhf(cval);
            out[((size_t)n * C16 + c) * SPAT + s]           = hval;
            out[outHalf + ((size_t)n * C16 + c) * SPAT + s] = cval;
        }
    }
}

extern "C" void kernel_launch(void* const* d_in, const int* in_sizes, int n_in,
                              void* d_out, int out_size, void* d_ws, size_t ws_size,
                              hipStream_t stream) {
    const float* mh  = (const float*)d_in[0];   // mailbox_h [128,8,16,16,16,16]
    const float* mc  = (const float*)d_in[1];   // mailbox_c
    const float* ufw = (const float*)d_in[2];   // U_f_w   [16,16,3,3,3]
    const float* uiw = (const float*)d_in[3];   // U_iou_w [48,16,3,3,3]
    const float* bio = (const float*)d_in[4];   // b_iou   [48,16,16,16]
    float* out = (float*)d_out;

    _Float16* hchild = (_Float16*)d_ws;                               // 16 MB
    float* csum = (float*)((char*)d_ws +
                           (size_t)NB * SPAT * C16 * sizeof(_Float16)); // +32 MB

    k_fconv_csum<<<NB * 4, 512, 0, stream>>>(mh, mc, ufw, csum, hchild);
    k_iou       <<<NB * 4, 256, 0, stream>>>(hchild, uiw, bio, csum, out);
}
